// MUTAGModel_54202487275652
// MI455X (gfx1250) — matrix operand order repaired and checked by reading, not compiled, not run on hardware
//
#include <hip/hip_runtime.h>
#include <math.h>

typedef _Float16 v16h __attribute__((ext_vector_type(16)));
typedef _Float16 v8h  __attribute__((ext_vector_type(8)));
typedef float    v8f  __attribute__((ext_vector_type(8)));

// ---------------- small utility kernels ----------------

__global__ void zero_f32(float* __restrict__ p, long long n) {
    long long i = (long long)blockIdx.x * blockDim.x + threadIdx.x;
    if (i < n) p[i] = 0.0f;
}

__global__ void deg_init(float* __restrict__ deg, int n) {
    int i = blockIdx.x * blockDim.x + threadIdx.x;
    if (i < n) deg[i] = 1.0f;   // self-loop contributes 1 to every node's degree
}

__global__ void deg_count(const int* __restrict__ dst, float* __restrict__ deg, int e) {
    int i = blockIdx.x * blockDim.x + threadIdx.x;
    if (i < e) atomicAdd(&deg[dst[i]], 1.0f);
}

__global__ void deg_finalize(float* __restrict__ deg, int n) {
    int i = blockIdx.x * blockDim.x + threadIdx.x;
    if (i < n) {
        float d = deg[i];
        deg[i] = (d > 0.0f) ? rsqrtf(d) : 0.0f;   // becomes deg_inv_sqrt
    }
}

// Pre-swizzle W (128x128 f32, row-major [k][c]) into per-lane WMMA B fragments:
// Wfrag[((ntile*4 + kc)*32 + lane)*16 + j] = (f16) W[(kc*32 + (lane>>4)*16 + j)*128
//                                                    + ntile*16 + (lane&15)]
// so a lane's fragment for (ntile, kc) is 16 contiguous halfs (32 B).
__global__ void cvt_w_frag(const float* __restrict__ W, _Float16* __restrict__ Wfrag) {
    int gid = blockIdx.x * blockDim.x + threadIdx.x;   // 0 .. 16383
    if (gid >= 128 * 128) return;
    int j     = gid & 15;
    int lane  = (gid >> 4) & 31;
    int kc    = (gid >> 9) & 3;
    int ntile = gid >> 11;
    int k   = kc * 32 + (lane >> 4) * 16 + j;
    int col = ntile * 16 + (lane & 15);
    Wfrag[gid] = (_Float16)W[k * 128 + col];
}

// out[n][c] = sum_k x[n][k] * W1[k][c]   (K = 7, C = 128)
__global__ void gemm_in7(const float* __restrict__ x, const float* __restrict__ W1,
                         float* __restrict__ out, int nrows) {
    long long gid = (long long)blockIdx.x * blockDim.x + threadIdx.x;
    if (gid >= (long long)nrows * 128) return;
    int n = (int)(gid >> 7);
    int c = (int)(gid & 127);
    const float* xr = x + (size_t)n * 7;
    float acc = 0.0f;
#pragma unroll
    for (int k = 0; k < 7; ++k) acc = fmaf(xr[k], W1[k * 128 + c], acc);
    out[gid] = acc;
}

// ---------------- WMMA GEMM: Out[N x 128] = H[N x 128] @ W[128 x 128] ----------------
// block = 256 threads = 8 waves; each block handles a 16-row strip, each wave one
// 16-column tile.  K = 128 swept in 4 chunks of 32 via v_wmma_f32_16x16x32_f16.
// B fragments come straight from the pre-swizzled global Wfrag (L2-resident, 32 KB);
// only the 16x128 A tile (4 KB f16) is staged in LDS.
__global__ void __launch_bounds__(256)
gemm128_wmma(const float* __restrict__ H, const _Float16* __restrict__ Wfrag,
             float* __restrict__ Out, int nrows) {
    __shared__ _Float16 sA[16 * 128];    // 4 KB

    const int tid  = threadIdx.x;
    const int wave = tid >> 5;           // 0..7 -> column tile
    const int lane = tid & 31;

    // hoist this wave's 4 B fragments (overlaps the LDS staging below)
    const v16h* WB = (const v16h*)Wfrag;
    v16h b0 = WB[(wave * 4 + 0) * 32 + lane];
    v16h b1 = WB[(wave * 4 + 1) * 32 + lane];
    v16h b2 = WB[(wave * 4 + 2) * 32 + lane];
    v16h b3 = WB[(wave * 4 + 3) * 32 + lane];

    // stage 16x128 A tile, f32 -> f16
    const int rowBase = blockIdx.x * 16;
    const bool fullTile = (rowBase + 16 <= nrows);   // uniform across block
    for (int i = tid; i < 16 * 128; i += 256) {
        int r = i >> 7, c = i & 127;
        float v = (fullTile || rowBase + r < nrows)
                      ? H[(size_t)(rowBase + r) * 128 + c] : 0.0f;
        sA[i] = (_Float16)v;
    }
    __syncthreads();

    const int m  = lane & 15;            // A row / B,C,D column within tile
    const int hi = lane >> 4;            // lane half select
    const int ncol = wave * 16 + m;      // global column 0..127

    // A fragment loader: lanes 0-15 K = {0..15}, lanes 16-31 K = {16..31}
    const _Float16* arow = sA + m * 128 + hi * 16;
    v8f acc = {};
#pragma unroll
    for (int kc = 0; kc < 4; ++kc) {
        const v8h* ap = (const v8h*)(arow + kc * 32);
        v8h alo = ap[0];                 // ds_load_b128
        v8h ahi = ap[1];                 // ds_load_b128 (K base+8..15)
        v16h a;
#pragma unroll
        for (int j = 0; j < 8; ++j) { a[j] = alo[j]; a[8 + j] = ahi[j]; }
        const v16h b = (kc == 0) ? b0 : (kc == 1) ? b1 : (kc == 2) ? b2 : b3;
        acc = __builtin_amdgcn_wmma_f32_16x16x32_f16(
            /*neg_a=*/false, a, /*neg_b=*/false, b,
            /*c_mod=*/(short)0, acc, /*reuse_a=*/false, /*reuse_b=*/false);
    }

    // D layout: VGPR i holds row M = i + 8*hi.  Uniform scalar branch: full tiles
    // (all of them when nrows % 16 == 0) take the straight-line store path.
    float* o = Out + (size_t)(rowBase + hi * 8) * 128 + ncol;
    if (fullTile) {
#pragma unroll
        for (int i = 0; i < 8; ++i) o[(size_t)i * 128] = acc[i];
    } else {
#pragma unroll
        for (int i = 0; i < 8; ++i)
            if (rowBase + hi * 8 + i < nrows) o[(size_t)i * 128] = acc[i];
    }
}

// ---------------- edge scatter: Out[dst] += T[src] * dinv[src]*dinv[dst] ----------------
// one wave per edge, each lane handles 4 contiguous columns (float4 gather + 4 atomics)
__global__ void scatter_edges(const float* __restrict__ T, const int* __restrict__ src,
                              const int* __restrict__ dst, const float* __restrict__ dinv,
                              float* __restrict__ Out, long long E, long long Ntot) {
    long long gid = (long long)blockIdx.x * blockDim.x + threadIdx.x;
    long long total = (E + Ntot) << 5;
    if (gid >= total) return;
    long long e = gid >> 5;
    int c4 = (int)(gid & 31) * 4;
    int s, d;
    if (e < E) { s = src[e]; d = dst[e]; }
    else       { s = d = (int)(e - E); }       // self-loop
    float nrm = dinv[s] * dinv[d];
    float4 v = *(const float4*)(T + (size_t)s * 128 + c4);
    float* o = Out + (size_t)d * 128 + c4;
    atomicAdd(o + 0, v.x * nrm);
    atomicAdd(o + 1, v.y * nrm);
    atomicAdd(o + 2, v.z * nrm);
    atomicAdd(o + 3, v.w * nrm);
}

__global__ void bias_relu(float* __restrict__ A, const float* __restrict__ b, long long total) {
    long long gid = (long long)blockIdx.x * blockDim.x + threadIdx.x;
    if (gid >= total) return;
    float v = A[gid] + b[gid & 127];
    A[gid] = fmaxf(v, 0.0f);
}

// segment-max pool; values are >= 0 post-ReLU so unsigned-bit atomicMax is order-preserving
__global__ void pool_max(const float* __restrict__ H, const int* __restrict__ batch,
                         float* __restrict__ pooled, long long total) {
    long long gid = (long long)blockIdx.x * blockDim.x + threadIdx.x;
    if (gid >= total) return;
    int i = (int)(gid >> 7);
    int c = (int)(gid & 127);
    int g = batch[i];
    atomicMax((unsigned int*)&pooled[(size_t)g * 128 + c], __float_as_uint(H[gid]));
}

// out[g] = elu(dot(pooled[g], Wl) + bl)
__global__ void head_kernel(const float* __restrict__ pooled, const float* __restrict__ Wl,
                            const float* __restrict__ bl, float* __restrict__ out) {
    __shared__ float red[128];
    int g = blockIdx.x, c = threadIdx.x;
    red[c] = pooled[(size_t)g * 128 + c] * Wl[c];
    __syncthreads();
    for (int s = 64; s > 0; s >>= 1) {
        if (c < s) red[c] += red[c + s];
        __syncthreads();
    }
    if (c == 0) {
        float v = red[0] + bl[0];
        out[g] = (v > 0.0f) ? v : (expf(v) - 1.0f);
    }
}

// ---------------- launcher ----------------

static inline unsigned int blocks_for(long long n, int bs) {
    return (unsigned int)((n + bs - 1) / bs);
}

extern "C" void kernel_launch(void* const* d_in, const int* in_sizes, int n_in,
                              void* d_out, int out_size, void* d_ws, size_t ws_size,
                              hipStream_t stream) {
    const float* x     = (const float*)d_in[0];
    const int*   edges = (const int*)d_in[1];
    const int*   batch = (const int*)d_in[2];
    const float* W1    = (const float*)d_in[3];
    const float* b1    = (const float*)d_in[4];
    const float* W2    = (const float*)d_in[5];
    const float* b2    = (const float*)d_in[6];
    const float* W3    = (const float*)d_in[7];
    const float* b3    = (const float*)d_in[8];
    const float* Wl    = (const float*)d_in[9];
    const float* bl    = (const float*)d_in[10];
    float* out = (float*)d_out;

    const int N = in_sizes[0] / 7;        // 200000
    const int E = in_sizes[1] / 2;        // 3200000
    const int G = out_size;               // 1000
    const int* src = edges;
    const int* dst = edges + E;

    // workspace carve-up
    char* ws = (char*)d_ws;
    size_t off = 0;
    float* dinv = (float*)(ws + off);  off += (((size_t)N * 4) + 255) & ~(size_t)255;
    float* bufA = (float*)(ws + off);  off += (size_t)N * 128 * 4;   // h / aggregate
    float* bufB = (float*)(ws + off);  off += (size_t)N * 128 * 4;   // transformed
    _Float16* Wfrag = (_Float16*)(ws + off); off += (size_t)128 * 128 * 2;
    float* pooled = (float*)(ws + off);      off += (size_t)G * 128 * 4;

    const long long NC   = (long long)N * 128;
    const long long SCAT = ((long long)E + N) << 5;

    // symmetric normalization terms
    deg_init    <<<blocks_for(N, 256), 256, 0, stream>>>(dinv, N);
    deg_count   <<<blocks_for(E, 256), 256, 0, stream>>>(dst, dinv, E);
    deg_finalize<<<blocks_for(N, 256), 256, 0, stream>>>(dinv, N);

    // ---- layer 1: x(Nx7) @ W1 -> scatter -> +b1, relu ----
    gemm_in7     <<<blocks_for(NC, 256), 256, 0, stream>>>(x, W1, bufB, N);
    zero_f32     <<<blocks_for(NC, 256), 256, 0, stream>>>(bufA, NC);
    scatter_edges<<<blocks_for(SCAT, 256), 256, 0, stream>>>(bufB, src, dst, dinv, bufA, E, N);
    bias_relu    <<<blocks_for(NC, 256), 256, 0, stream>>>(bufA, b1, NC);

    // ---- layers 2 and 3: WMMA 128x128 transform -> scatter -> bias+relu ----
    const float* Ws[2] = { W2, W3 };
    const float* bs[2] = { b2, b3 };
    for (int l = 0; l < 2; ++l) {
        cvt_w_frag  <<<blocks_for(128 * 128, 256), 256, 0, stream>>>(Ws[l], Wfrag);
        gemm128_wmma<<<blocks_for(N, 16), 256, 0, stream>>>(bufA, Wfrag, bufB, N);
        zero_f32    <<<blocks_for(NC, 256), 256, 0, stream>>>(bufA, NC);
        scatter_edges<<<blocks_for(SCAT, 256), 256, 0, stream>>>(bufB, src, dst, dinv, bufA, E, N);
        bias_relu   <<<blocks_for(NC, 256), 256, 0, stream>>>(bufA, bs[l], NC);
    }

    // ---- pooling + head ----
    zero_f32   <<<blocks_for((long long)G * 128, 256), 256, 0, stream>>>(pooled, (long long)G * 128);
    pool_max   <<<blocks_for(NC, 256), 256, 0, stream>>>(bufA, batch, pooled, NC);
    head_kernel<<<G, 128, 0, stream>>>(pooled, Wl, bl, out);
}
